// Step_13322988552635
// MI455X (gfx1250) — matrix operand order repaired and checked by reading, not compiled, not run on hardware
//
#include <hip/hip_runtime.h>
#include <math.h>

// Problem constants
#define BB 2
#define SS 4096
#define DD 2048
#define MM (BB * SS)            // 8192 rows of the GEMM
#define SEG ((size_t)BB * SS * DD)  // 16,777,216 floats per output tensor
#define NCH 32                  // scan chunks
#define CHL (SS / NCH)          // 128 steps per chunk
#define NCOL (BB * DD)          // 4096 independent scan columns

typedef __attribute__((ext_vector_type(16))) __bf16 v16bf;
typedef __attribute__((ext_vector_type(8)))  float  v8f;

static __device__ __forceinline__ unsigned short f32_to_bf16(float f) {
    unsigned int u = __float_as_uint(f);
    unsigned int r = u + 0x7FFFu + ((u >> 16) & 1u);   // round-to-nearest-even
    return (unsigned short)(r >> 16);
}

// ---------------------------------------------------------------------------
// K0: f32 -> bf16 convert (weights)
// ---------------------------------------------------------------------------
__global__ void __launch_bounds__(256) cvt_bf16_kernel(const float* __restrict__ src,
                                                       unsigned short* __restrict__ dst) {
    size_t i = (size_t)blockIdx.x * blockDim.x + threadIdx.x;   // exact grid
    dst[i] = f32_to_bf16(src[i]);
}

// ---------------------------------------------------------------------------
// K1: gate_input = rms_norm(c) + rms_norm(prev_fetched), stored as bf16.
// One block (256 thr, 8 waves) per row of D=2048.
// ---------------------------------------------------------------------------
__global__ void __launch_bounds__(256) gate_kernel(const float* __restrict__ c,
                                                   const float* __restrict__ pf,
                                                   unsigned short* __restrict__ gate) {
    const size_t row = blockIdx.x;
    const float* cr = c  + row * DD;
    const float* pr = pf + row * DD;
    float sc = 0.f, sp = 0.f;
    for (int i = threadIdx.x; i < DD; i += 256) {
        float a = cr[i]; sc = fmaf(a, a, sc);
        float b = pr[i]; sp = fmaf(b, b, sp);
    }
    // wave32 reduce
    for (int off = 16; off > 0; off >>= 1) {
        sc += __shfl_down(sc, off, 32);
        sp += __shfl_down(sp, off, 32);
    }
    __shared__ float redc[8], redp[8];
    if ((threadIdx.x & 31) == 0) {
        redc[threadIdx.x >> 5] = sc;
        redp[threadIdx.x >> 5] = sp;
    }
    __syncthreads();
    float tc = 0.f, tp = 0.f;
    #pragma unroll
    for (int w = 0; w < 8; ++w) { tc += redc[w]; tp += redp[w]; }
    const float rc = rsqrtf(tc * (1.0f / DD) + 1e-6f);
    const float rp = rsqrtf(tp * (1.0f / DD) + 1e-6f);
    unsigned short* gr = gate + row * DD;
    for (int i = threadIdx.x; i < DD; i += 256)
        gr[i] = f32_to_bf16(fmaf(cr[i], rc, pr[i] * rp));
}

// ---------------------------------------------------------------------------
// K2: fused dual GEMM + activation epilogue.
//   acc_a = gate @ W_alpha ; acc_b = gate @ W_beta     (bf16 WMMA, f32 acc)
//   alphas = sigmoid(acc_a + b_alpha)  -> alph_out
//   x      = v * silu(acc_b + b_beta) * sqrt(max(1-a^2,1e-6)) -> x_out
// Block: 256 thr = 8 waves arranged 2(M) x 4(N); wave tile 64(M) x 32(N);
// block tile 128 x 128.  Per K=32 step: 8 fragment loads feed 16 WMMAs.
// ---------------------------------------------------------------------------
__global__ void __launch_bounds__(256) gemm_gate_kernel(
    const unsigned short* __restrict__ G,    // [MM][DD] bf16
    const unsigned short* __restrict__ Wa,   // [DD][DD] bf16
    const unsigned short* __restrict__ Wb,   // [DD][DD] bf16
    const float* __restrict__ b_alpha,
    const float* __restrict__ b_beta,
    const float* __restrict__ v_in,
    float* __restrict__ alph_out,
    float* __restrict__ x_out) {

    const int lane = threadIdx.x & 31;
    const int wave = threadIdx.x >> 5;
    const int wm = wave >> 2;                  // 0..1
    const int wn = wave & 3;                   // 0..3
    const int m_wave = blockIdx.y * 128 + wm * 64;
    const int n_wave = blockIdx.x * 128 + wn * 32;
    const int half = lane >> 4;
    const int l16  = lane & 15;

    v8f acc_a[4][2], acc_b[4][2];
    #pragma unroll
    for (int i = 0; i < 4; ++i)
        #pragma unroll
        for (int j = 0; j < 2; ++j) { acc_a[i][j] = (v8f)0.0f; acc_b[i][j] = (v8f)0.0f; }

    for (int k0 = 0; k0 < DD; k0 += 32) {
        // A fragments: lane holds row m = l16, K interleave per ISA 16-bit layout
        v16bf af[4];
        #pragma unroll
        for (int i = 0; i < 4; ++i) {
            const unsigned short* p =
                G + (size_t)(m_wave + i * 16 + l16) * DD + k0 + half * 8;
            union { uint4 u[2]; v16bf v; } ua;
            ua.u[0] = *(const uint4*)(p);
            ua.u[1] = *(const uint4*)(p + 16);
            af[i] = ua.v;
        }
        // B fragments: lane holds column n = l16, K interleave per ISA 16-bit layout
        v16bf ba[2], bbf[2];
        #pragma unroll
        for (int j = 0; j < 2; ++j) {
            const unsigned short* pa = Wa + (size_t)k0 * DD + n_wave + j * 16 + l16;
            const unsigned short* pb = Wb + (size_t)k0 * DD + n_wave + j * 16 + l16;
            union { unsigned short h[16]; v16bf v; } u1, u2;
            #pragma unroll
            for (int s = 0; s < 16; ++s) {
                const int kk = half * 8 + (s & 7) + (s >> 3) * 16;
                u1.h[s] = pa[(size_t)kk * DD];
                u2.h[s] = pb[(size_t)kk * DD];
            }
            ba[j]  = u1.v;
            bbf[j] = u2.v;
        }
        #pragma unroll
        for (int i = 0; i < 4; ++i)
            #pragma unroll
            for (int j = 0; j < 2; ++j) {
                acc_a[i][j] = __builtin_amdgcn_wmma_f32_16x16x32_bf16(
                    false, af[i], false, ba[j],  (short)0, acc_a[i][j], false, false);
                acc_b[i][j] = __builtin_amdgcn_wmma_f32_16x16x32_bf16(
                    false, af[i], false, bbf[j], (short)0, acc_b[i][j], false, false);
            }
    }

    // Epilogue (C layout: n = l16, VGPR r -> m = r + 8*half)
    #pragma unroll
    for (int i = 0; i < 4; ++i) {
        #pragma unroll
        for (int j = 0; j < 2; ++j) {
            const int ncol = n_wave + j * 16 + l16;
            const float biasA = b_alpha[ncol];
            const float biasB = b_beta[ncol];
            #pragma unroll
            for (int r = 0; r < 8; ++r) {
                const int mrow = m_wave + i * 16 + half * 8 + r;
                const size_t idx = (size_t)mrow * DD + ncol;
                const float ta = acc_a[i][j][r] + biasA;
                const float tb = acc_b[i][j][r] + biasB;
                const float al = 1.0f / (1.0f + __expf(-ta));     // sigmoid
                const float be = tb / (1.0f + __expf(-tb));       // silu
                const float wsc = sqrtf(fmaxf(1.0f - al * al, 1e-6f));
                alph_out[idx] = al;
                x_out[idx]    = v_in[idx] * be * wsc;
            }
        }
    }
}

// ---------------------------------------------------------------------------
// K3: chunk-local scan.  h_t = a_t*h_{t-1} + x_t within each 128-step chunk.
// Writes local h in place of x; emits per-chunk (h_end, prod_a) carries.
// ---------------------------------------------------------------------------
__global__ void __launch_bounds__(256) scan_chunk_kernel(
    const float* __restrict__ alph, float* __restrict__ hbuf,
    float* __restrict__ Hend, float* __restrict__ Aprod) {
    const int tid = blockIdx.x * blockDim.x + threadIdx.x;   // exact: NCOL*NCH
    const int d  = tid & (DD - 1);
    const int t2 = tid >> 11;
    const int b  = t2 & (BB - 1);
    const int ch = t2 >> 1;
    const size_t base = ((size_t)b * SS + (size_t)ch * CHL) * DD + d;
    float h = 0.f, P = 1.f;
    for (int t = 0; t < CHL; ++t) {
        const size_t idx = base + (size_t)t * DD;
        const float a = alph[idx];
        const float x = hbuf[idx];
        h = fmaf(a, h, x);
        P *= a;
        hbuf[idx] = h;
    }
    const int col = b * DD + d;
    Hend[ch * NCOL + col]  = h;
    Aprod[ch * NCOL + col] = P;
}

// ---------------------------------------------------------------------------
// K4: serial scan over the 32 chunk carries per column (cheap).
// ---------------------------------------------------------------------------
__global__ void __launch_bounds__(256) scan_carry_kernel(
    const float* __restrict__ Hend, const float* __restrict__ Aprod,
    float* __restrict__ CarryIn) {
    const int col = blockIdx.x * blockDim.x + threadIdx.x;   // exact: NCOL
    float c = 0.f;
    for (int ch = 0; ch < NCH; ++ch) {
        CarryIn[ch * NCOL + col] = c;
        c = fmaf(Aprod[ch * NCOL + col], c, Hend[ch * NCOL + col]);
    }
}

// ---------------------------------------------------------------------------
// K5: apply carries + final epilogue:
//   fetched = h_local + prefixprod(a)*carry ;  out_new = out + f*g_out ;
//   v_new = v + f*g_v
// ---------------------------------------------------------------------------
__global__ void __launch_bounds__(256) scan_apply_kernel(
    const float* __restrict__ alph, float* __restrict__ fet,
    const float* __restrict__ CarryIn,
    const float* __restrict__ v_in, const float* __restrict__ out_in,
    const float* __restrict__ g_out, const float* __restrict__ g_v,
    float* __restrict__ v_new, float* __restrict__ out_new) {
    const int tid = blockIdx.x * blockDim.x + threadIdx.x;
    const int d  = tid & (DD - 1);
    const int t2 = tid >> 11;
    const int b  = t2 & (BB - 1);
    const int ch = t2 >> 1;
    const float carry = CarryIn[ch * NCOL + b * DD + d];
    const float go = g_out[d];
    const float gv = g_v[d];
    const size_t base = ((size_t)b * SS + (size_t)ch * CHL) * DD + d;
    float P = 1.f;
    for (int t = 0; t < CHL; ++t) {
        const size_t idx = base + (size_t)t * DD;
        P *= alph[idx];
        const float f = fmaf(P, carry, fet[idx]);
        fet[idx]     = f;
        out_new[idx] = fmaf(f, go, out_in[idx]);
        v_new[idx]   = fmaf(f, gv, v_in[idx]);
    }
}

// ---------------------------------------------------------------------------
// K6: vectorized copy (c passthrough)
// ---------------------------------------------------------------------------
__global__ void __launch_bounds__(256) copy4_kernel(const float4* __restrict__ src,
                                                    float4* __restrict__ dst) {
    const size_t i = (size_t)blockIdx.x * blockDim.x + threadIdx.x;   // exact
    dst[i] = src[i];
}

// ---------------------------------------------------------------------------
extern "C" void kernel_launch(void* const* d_in, const int* in_sizes, int n_in,
                              void* d_out, int out_size, void* d_ws, size_t ws_size,
                              hipStream_t stream) {
    const float* v_in   = (const float*)d_in[0];
    const float* out_in = (const float*)d_in[1];
    const float* c_in   = (const float*)d_in[2];
    const float* pf_in  = (const float*)d_in[3];
    const float* W_a    = (const float*)d_in[4];
    const float* b_a    = (const float*)d_in[5];
    const float* W_b    = (const float*)d_in[6];
    const float* b_b    = (const float*)d_in[7];
    const float* g_out  = (const float*)d_in[8];
    const float* g_v    = (const float*)d_in[9];

    float* out = (float*)d_out;
    float* seg0 = out;              // v_new     (gate_bf16 scratch early)
    float* seg1 = out + SEG;        // out_new   (Wa/Wb bf16 scratch early)
    float* seg2 = out + 2 * SEG;    // c         (alphas scratch early)
    float* seg3 = out + 3 * SEG;    // fetched   (x / local-h in flight)

    // bf16 staging inside not-yet-final output segments
    unsigned short* gate16 = (unsigned short*)seg0;                 // 32 MB of 64
    unsigned short* Wa16   = (unsigned short*)seg1;                 //  8 MB
    unsigned short* Wb16   = Wa16 + (size_t)DD * DD;                //  8 MB

    // scan carries in ws (~1.5 MB)
    float* Hend    = (float*)d_ws;
    float* Aprod   = Hend  + (size_t)NCH * NCOL;
    float* CarryIn = Aprod + (size_t)NCH * NCOL;

    const int wElems = DD * DD;                        // 4,194,304
    cvt_bf16_kernel<<<wElems / 256, 256, 0, stream>>>(W_a, Wa16);
    cvt_bf16_kernel<<<wElems / 256, 256, 0, stream>>>(W_b, Wb16);

    gate_kernel<<<MM, 256, 0, stream>>>(c_in, pf_in, gate16);

    dim3 gemmGrid(DD / 128, MM / 128);                 // (16, 64)
    gemm_gate_kernel<<<gemmGrid, 256, 0, stream>>>(gate16, Wa16, Wb16,
                                                   b_a, b_b, v_in,
                                                   /*alphas*/ seg2, /*x*/ seg3);

    scan_chunk_kernel<<<(NCOL * NCH) / 256, 256, 0, stream>>>(seg2, seg3, Hend, Aprod);
    scan_carry_kernel<<<NCOL / 256, 256, 0, stream>>>(Hend, Aprod, CarryIn);
    scan_apply_kernel<<<(NCOL * NCH) / 256, 256, 0, stream>>>(seg2, seg3, CarryIn,
                                                              v_in, out_in, g_out, g_v,
                                                              /*v_new*/ seg0,
                                                              /*out_new*/ seg1);

    copy4_kernel<<<(SEG / 4) / 256, 256, 0, stream>>>((const float4*)c_in, (float4*)seg2);
}
